// NicheTrans_ct_58703613002251
// MI455X (gfx1250) — matrix operand order repaired and checked by reading, not compiled, not run on hardware
//
#include <hip/hip_runtime.h>
#include <hip/hip_bf16.h>

typedef __attribute__((ext_vector_type(16))) _Float16 v16h;
typedef __attribute__((ext_vector_type(8)))  float    v8f;

#define EPS 1e-5f
#define BNS 0.9999950000375f   /* 1/sqrt(1+1e-5) */

__device__ __forceinline__ float wave_sum(float x) {
  #pragma unroll
  for (int m = 16; m > 0; m >>= 1) x += __shfl_xor(x, m, 32);
  return x;
}
__device__ __forceinline__ float leaky(float x) { return x > 0.f ? x : 0.01f * x; }

// ---------------------------------------------------------------------------
// Generic f16 GEMM: C(f32, MxN) = A(f16, MxK, lda) * B(f16, KxN, ldb)
// 64x64 block tile, BK=64, 128 threads = 4 waves, each wave computes 16x64.
// Per LDS stage: 2 K-substeps x 4 N-tiles = 8 WMMAs per wave.
// sA: row-major, stride 72 halfs (144B, 16B aligned).
// sB: transposed (n-major), stride 72 halfs, with 32B-chunk XOR swizzle
//     (chunk c=k/16 stored at c ^ (n>>4)) for bank-conflict-free pair stores.
// ---------------------------------------------------------------------------
__global__ __launch_bounds__(128) void gemm16(const _Float16* __restrict__ A,
                                              const _Float16* __restrict__ Bm,
                                              float* __restrict__ C,
                                              int M, int N, int K,
                                              int lda, int ldb, int ldc) {
  (void)M; (void)N;
  __shared__ _Float16 sA[64 * 72];
  __shared__ _Float16 sB[64 * 72];
  const int tid  = threadIdx.x;
  const int wave = tid >> 5;
  const int lane = tid & 31;
  const int g    = lane >> 4;
  const int lm   = lane & 15;
  const int mBase = blockIdx.y * 64;
  const int nBase = blockIdx.x * 64;

  v8f acc0 = {}, acc1 = {}, acc2 = {}, acc3 = {};

  for (int k0 = 0; k0 < K; k0 += 64) {
    __syncthreads();
    {   // A tile: 64 rows x 64 halfs, 32 halfs per thread (4x b128)
      int row = tid >> 1;
      int col = (tid & 1) * 32;
      const _Float16* gp = A + (size_t)(mBase + row) * lda + k0 + col;
      uint4 d0 = *(const uint4*)(gp);
      uint4 d1 = *(const uint4*)(gp + 8);
      uint4 d2 = *(const uint4*)(gp + 16);
      uint4 d3 = *(const uint4*)(gp + 24);
      if (k0 + 64 < K) __builtin_prefetch(gp + 64, 0, 1);
      uint4* sp = (uint4*)&sA[row * 72 + col];
      sp[0] = d0; sp[1] = d1; sp[2] = d2; sp[3] = d3;
    }
    {   // B tile: 64(k) x 64(n); thread owns k-pair (2p,2p+1) x 16 n
        // store transposed as packed b32 (k,k+1) with chunk swizzle
      int p  = tid >> 2;            // 0..31  -> k = 2p, 2p+1
      int nc = (tid & 3) * 16;      // n chunk base
      int c  = p >> 3;              // k 32B-chunk index 0..3
      int kk = 2 * (p & 7);         // k offset within chunk
      const _Float16* g0 = Bm + (size_t)(k0 + 2 * p) * ldb + nBase + nc;
      const _Float16* g1 = g0 + ldb;
      uint4 a0 = *(const uint4*)(g0);
      uint4 a1 = *(const uint4*)(g0 + 8);
      uint4 b0 = *(const uint4*)(g1);
      uint4 b1 = *(const uint4*)(g1 + 8);
      const unsigned short* r0  = (const unsigned short*)&a0;
      const unsigned short* r0b = (const unsigned short*)&a1;
      const unsigned short* r1  = (const unsigned short*)&b0;
      const unsigned short* r1b = (const unsigned short*)&b1;
      #pragma unroll
      for (int j = 0; j < 8; ++j) {
        int n  = nc + j;
        int sw = (c ^ (n >> 4)) * 16;
        *(unsigned*)&sB[n * 72 + sw + kk] = (unsigned)r0[j] | ((unsigned)r1[j] << 16);
      }
      #pragma unroll
      for (int j = 0; j < 8; ++j) {
        int n  = nc + 8 + j;
        int sw = (c ^ (n >> 4)) * 16;
        *(unsigned*)&sB[n * 72 + sw + kk] = (unsigned)r0b[j] | ((unsigned)r1b[j] << 16);
      }
    }
    __syncthreads();

    int arow = wave * 16 + lm;
    #pragma unroll
    for (int s = 0; s < 2; ++s) {
      // A fragment: lane row M=wave*16+lm; halfs i: K = s*32 + 16g + i
      union { v16h v; uint4 q[2]; } fa;
      fa.q[0] = *(const uint4*)(&sA[arow * 72 + s * 32 + g * 16]);
      fa.q[1] = *(const uint4*)(&sA[arow * 72 + s * 32 + g * 16 + 8]);
      // B fragments: lane col N=nt*16+lm; halfs i: K = s*32 + 16g + i
      // stored chunk (2s+g) lives at swizzled chunk (2s+g)^nt
      union { v16h v; uint4 q[2]; } fb0, fb1, fb2, fb3;
      {
        const _Float16* bp = &sB[(0 * 16 + lm) * 72 + (((s * 2 + g) ^ 0) * 16)];
        fb0.q[0] = *(const uint4*)(bp);
        fb0.q[1] = *(const uint4*)(bp + 8);
      }
      {
        const _Float16* bp = &sB[(1 * 16 + lm) * 72 + (((s * 2 + g) ^ 1) * 16)];
        fb1.q[0] = *(const uint4*)(bp);
        fb1.q[1] = *(const uint4*)(bp + 8);
      }
      {
        const _Float16* bp = &sB[(2 * 16 + lm) * 72 + (((s * 2 + g) ^ 2) * 16)];
        fb2.q[0] = *(const uint4*)(bp);
        fb2.q[1] = *(const uint4*)(bp + 8);
      }
      {
        const _Float16* bp = &sB[(3 * 16 + lm) * 72 + (((s * 2 + g) ^ 3) * 16)];
        fb3.q[0] = *(const uint4*)(bp);
        fb3.q[1] = *(const uint4*)(bp + 8);
      }
      acc0 = __builtin_amdgcn_wmma_f32_16x16x32_f16(false, fa.v, false, fb0.v, (short)0, acc0, false, false);
      acc1 = __builtin_amdgcn_wmma_f32_16x16x32_f16(false, fa.v, false, fb1.v, (short)0, acc1, false, false);
      acc2 = __builtin_amdgcn_wmma_f32_16x16x32_f16(false, fa.v, false, fb2.v, (short)0, acc2, false, false);
      acc3 = __builtin_amdgcn_wmma_f32_16x16x32_f16(false, fa.v, false, fb3.v, (short)0, acc3, false, false);
    }
  }

  // C layout: VGPR r -> M = r + 8*g, N = lm
  #pragma unroll
  for (int r = 0; r < 8; ++r) {
    size_t m = (size_t)(mBase + wave * 16 + r + 8 * g);
    float* cp = C + m * ldc + nBase + lm;
    cp[0]  = acc0[r];
    cp[16] = acc1[r];
    cp[32] = acc2[r];
    cp[48] = acc3[r];
  }
}

// ---------------------------------------------------------------------------
// Weight conversion kernels
// ---------------------------------------------------------------------------
__global__ void conv_plain(const float* __restrict__ s, _Float16* __restrict__ d, size_t n) {
  size_t i = (size_t)blockIdx.x * 256 + threadIdx.x;
  if (i < n) d[i] = (_Float16)s[i];
}
__global__ void conv_w1(const float* __restrict__ s, _Float16* __restrict__ d) {
  size_t i = (size_t)blockIdx.x * 256 + threadIdx.x;            // over 896*512
  if (i >= (size_t)896 * 512) return;
  int k = (int)(i / 512), n = (int)(i % 512);
  d[i] = (k < 877) ? (_Float16)s[(size_t)k * 512 + n] : (_Float16)0.f;
}
__global__ void conv_predw(const float* __restrict__ s, _Float16* __restrict__ d) {
  size_t i = (size_t)blockIdx.x * 256 + threadIdx.x;            // over 256*17536
  if (i >= (size_t)256 * 17536) return;
  int dd = (int)(i / 17536), c = (int)(i % 17536);
  int t = c >> 7, hh = c & 127;
  d[i] = (_Float16)s[((size_t)t * 256 + dd) * 128 + hh];        // (T,D,H) -> (D, T*H)
}

// ---------------------------------------------------------------------------
// prep: build x (f16, padded 877->896) + valid mask; role/class additive term
// ---------------------------------------------------------------------------
__global__ __launch_bounds__(256) void prep_x(const float* __restrict__ src,
                                              const float* __restrict__ srcn,
                                              _Float16* __restrict__ xh,
                                              float* __restrict__ valid) {
  int r = blockIdx.x;           // B*9 rows
  int b = r / 9, t = r % 9;
  const float* p = (t == 0) ? src + (size_t)b * 877
                            : srcn + ((size_t)b * 8 + (t - 1)) * 877;
  __shared__ float red[256];
  float a = 0.f;
  for (int i = threadIdx.x; i < 896; i += 256) {
    float v = (i < 877) ? p[i] : 0.f;
    a += fabsf(v);
    xh[(size_t)r * 896 + i] = (_Float16)v;
  }
  red[threadIdx.x] = a; __syncthreads();
  for (int s = 128; s > 0; s >>= 1) {
    if (threadIdx.x < s) red[threadIdx.x] += red[threadIdx.x + s];
    __syncthreads();
  }
  if (threadIdx.x == 0) valid[r] = (red[0] > 0.f) ? 1.f : 0.f;
}

__global__ __launch_bounds__(256) void prep_tokens(const float* __restrict__ cell_inf,
                                                   const float* __restrict__ tc,
                                                   const float* __restrict__ t1,
                                                   const float* __restrict__ t2,
                                                   const float* __restrict__ ctok,
                                                   const float* __restrict__ valid,
                                                   float* __restrict__ addterm) {
  int r = blockIdx.x, t = r % 9, d = threadIdx.x;
  float vm = valid[r];
  float role = (t == 0) ? tc[d] : ((t <= 4) ? t1[d] : t2[d]);
  const float* ci = cell_inf + (size_t)r * 13;
  float cls = 0.f;
  #pragma unroll
  for (int c = 0; c < 13; ++c) cls += ctok[c * 256 + d] * ci[c];
  addterm[(size_t)r * 256 + d] = (role + cls) * vm;
}

// ---------------------------------------------------------------------------
// elementwise epilogues
// ---------------------------------------------------------------------------
__global__ void ep_bn_leaky(const float* __restrict__ in, const float* __restrict__ bias,
                            const float* __restrict__ g, const float* __restrict__ bb,
                            _Float16* __restrict__ out, int ncols, size_t n) {
  size_t i = (size_t)blockIdx.x * 256 + threadIdx.x;
  if (i >= n) return;
  int c = (int)(i % ncols);
  float x = in[i] + bias[c];
  out[i] = (_Float16)leaky(x * (g[c] * BNS) + bb[c]);
}
__global__ void ep_bn_leaky_add(const float* __restrict__ in, const float* __restrict__ bias,
                                const float* __restrict__ g, const float* __restrict__ bb,
                                const float* __restrict__ add, _Float16* __restrict__ out,
                                int ncols, size_t n) {
  size_t i = (size_t)blockIdx.x * 256 + threadIdx.x;
  if (i >= n) return;
  int c = (int)(i % ncols);
  float x = in[i] + bias[c];
  out[i] = (_Float16)(leaky(x * (g[c] * BNS) + bb[c]) + add[i]);
}
__global__ void ep_gelu(const float* __restrict__ in, const float* __restrict__ bias,
                        _Float16* __restrict__ out, int ncols, size_t n) {
  size_t i = (size_t)blockIdx.x * 256 + threadIdx.x;
  if (i >= n) return;
  int c = (int)(i % ncols);
  float z = in[i] + bias[c];
  float t = 0.7978845608f * (z + 0.044715f * z * z * z);   // jax.nn.gelu (tanh approx)
  out[i] = (_Float16)(0.5f * z * (1.f + tanhf(t)));
}
__global__ void ep_res_add(const float* __restrict__ in, const float* __restrict__ bias,
                           float* __restrict__ fo, int ncols, size_t n) {
  size_t i = (size_t)blockIdx.x * 256 + threadIdx.x;
  if (i >= n) return;
  int c = (int)(i % ncols);
  fo[i] += in[i] + bias[c];
}
__global__ void extract_feat(const float* __restrict__ fo, _Float16* __restrict__ feat, int Btot) {
  int i = blockIdx.x * 256 + threadIdx.x;        // B*256
  if (i >= Btot * 256) return;
  int b = i >> 8, d = i & 255;
  feat[i] = (_Float16)fo[(size_t)b * 9 * 256 + d];
}

// ---------------------------------------------------------------------------
// LN row kernels (wave per row, 256 cols)
// ---------------------------------------------------------------------------
__global__ __launch_bounds__(256) void ep_ln_relu_ln(const float* __restrict__ in,
    const float* __restrict__ pb, const float* __restrict__ g1, const float* __restrict__ b1,
    const float* __restrict__ g2, const float* __restrict__ b2,
    float* __restrict__ fo, _Float16* __restrict__ hn, size_t nrows) {
  size_t row = (size_t)blockIdx.x * 8 + (threadIdx.x >> 5);
  int lane = threadIdx.x & 31;
  if (row >= nrows) return;
  const float* xr = in + row * 256;
  float x[8];
  #pragma unroll
  for (int j = 0; j < 8; ++j) { int c = lane + 32 * j; x[j] = xr[c] + pb[c]; }
  float s = 0.f;
  #pragma unroll
  for (int j = 0; j < 8; ++j) s += x[j];
  float m = wave_sum(s) * (1.f / 256.f);
  float v = 0.f;
  #pragma unroll
  for (int j = 0; j < 8; ++j) { float d = x[j] - m; v += d * d; }
  float inv = rsqrtf(wave_sum(v) * (1.f / 256.f) + EPS);
  float y[8];
  #pragma unroll
  for (int j = 0; j < 8; ++j) {
    int c = lane + 32 * j;
    float t = (x[j] - m) * inv * g1[c] + b1[c];
    y[j] = t > 0.f ? t : 0.f;
    fo[row * 256 + c] = y[j];
  }
  s = 0.f;
  #pragma unroll
  for (int j = 0; j < 8; ++j) s += y[j];
  m = wave_sum(s) * (1.f / 256.f);
  v = 0.f;
  #pragma unroll
  for (int j = 0; j < 8; ++j) { float d = y[j] - m; v += d * d; }
  inv = rsqrtf(wave_sum(v) * (1.f / 256.f) + EPS);
  #pragma unroll
  for (int j = 0; j < 8; ++j) {
    int c = lane + 32 * j;
    hn[row * 256 + c] = (_Float16)((y[j] - m) * inv * g2[c] + b2[c]);
  }
}

__global__ __launch_bounds__(256) void ep_res_ln(const float* __restrict__ in,
    const float* __restrict__ rb, const float* __restrict__ g2, const float* __restrict__ b2,
    float* __restrict__ fo, _Float16* __restrict__ hn, size_t nrows) {
  size_t row = (size_t)blockIdx.x * 8 + (threadIdx.x >> 5);
  int lane = threadIdx.x & 31;
  if (row >= nrows) return;
  float x[8];
  #pragma unroll
  for (int j = 0; j < 8; ++j) {
    int c = lane + 32 * j;
    float t = fo[row * 256 + c] + in[row * 256 + c] + rb[c];
    fo[row * 256 + c] = t;
    x[j] = t;
  }
  float s = 0.f;
  #pragma unroll
  for (int j = 0; j < 8; ++j) s += x[j];
  float m = wave_sum(s) * (1.f / 256.f);
  float v = 0.f;
  #pragma unroll
  for (int j = 0; j < 8; ++j) { float d = x[j] - m; v += d * d; }
  float inv = rsqrtf(wave_sum(v) * (1.f / 256.f) + EPS);
  #pragma unroll
  for (int j = 0; j < 8; ++j) {
    int c = lane + 32 * j;
    hn[row * 256 + c] = (_Float16)((x[j] - m) * inv * g2[c] + b2[c]);
  }
}

// ---------------------------------------------------------------------------
// attention: one block per batch element, one wave per head (N=9, H=4, DH=64)
// ---------------------------------------------------------------------------
__global__ __launch_bounds__(128) void attn_kernel(const float* __restrict__ q,
    const float* __restrict__ k, const float* __restrict__ v,
    const float* __restrict__ valid, _Float16* __restrict__ o16) {
  __shared__ float sq[9 * 256], sk[9 * 256], sv[9 * 256];
  __shared__ float ss[4][9][12];
  __shared__ float sval[9];
  int b = blockIdx.x, tid = threadIdx.x;
  size_t base = (size_t)b * 9 * 256;
  for (int i = tid; i < 9 * 256; i += 128) { sq[i] = q[base + i]; sk[i] = k[base + i]; sv[i] = v[base + i]; }
  if (tid < 9) sval[tid] = valid[b * 9 + tid];
  __syncthreads();
  int h = tid >> 5, lane = tid & 31;
  for (int p = lane; p < 81; p += 32) {
    int i = p / 9, j = p % 9;
    const float* qp = &sq[i * 256 + h * 64];
    const float* kp = &sk[j * 256 + h * 64];
    float s = 0.f;
    #pragma unroll 8
    for (int d = 0; d < 64; ++d) s += qp[d] * kp[d];
    ss[h][i][j] = (sval[j] > 0.f) ? s * 0.125f : -1e9f;
  }
  __syncthreads();
  for (int i = lane; i < 9; i += 32) {
    float mx = -3e38f;
    #pragma unroll
    for (int j = 0; j < 9; ++j) mx = fmaxf(mx, ss[h][i][j]);
    float sum = 0.f, e[9];
    #pragma unroll
    for (int j = 0; j < 9; ++j) { e[j] = __expf(ss[h][i][j] - mx); sum += e[j]; }
    float rs = 1.f / sum;
    #pragma unroll
    for (int j = 0; j < 9; ++j) ss[h][i][j] = e[j] * rs;
  }
  __syncthreads();
  for (int p = lane; p < 9 * 64; p += 32) {
    int i = p >> 6, d = p & 63;
    float acc = 0.f;
    #pragma unroll
    for (int j = 0; j < 9; ++j) acc += ss[h][i][j] * sv[j * 256 + h * 64 + d];
    o16[base + i * 256 + h * 64 + d] = (_Float16)acc;
  }
}

// ---------------------------------------------------------------------------
// prediction head reduction: out[b,t] = sum_h leaky(bn(ph+b1)) * w2
// ---------------------------------------------------------------------------
__global__ __launch_bounds__(256) void pred_reduce(const float* __restrict__ ph,
    const float* __restrict__ pb1, const float* __restrict__ bng, const float* __restrict__ bnb,
    const float* __restrict__ w2, float* __restrict__ out, int Btot) {
  int w = blockIdx.x * 8 + (threadIdx.x >> 5);
  if (w >= Btot) return;
  int lane = threadIdx.x & 31;
  int b = w / 137, t = w % 137;
  const float* pr = ph + (size_t)b * 17536 + t * 128;
  float s = 0.f;
  #pragma unroll
  for (int j = 0; j < 4; ++j) {
    int hh = lane + 32 * j, c = t * 128 + hh;
    float x = pr[hh] + pb1[c];
    s += leaky(x * (bng[c] * BNS) + bnb[c]) * w2[c];
  }
  s = wave_sum(s);
  if (lane == 0) out[(size_t)b * 137 + t] = s;
}

// ---------------------------------------------------------------------------
extern "C" void kernel_launch(void* const* d_in, const int* in_sizes, int n_in,
                              void* d_out, int out_size, void* d_ws, size_t ws_size,
                              hipStream_t stream) {
  (void)in_sizes; (void)n_in; (void)out_size; (void)ws_size;
  const int   Bb = 4096, T = 137;
  const size_t R = (size_t)Bb * 9;

  const float* source   = (const float*)d_in[0];
  const float* srcn     = (const float*)d_in[1];
  const float* cell_inf = (const float*)d_in[2];
  const float* enc_w1   = (const float*)d_in[3];
  const float* enc_b1   = (const float*)d_in[4];
  const float* bn1_g    = (const float*)d_in[5];
  const float* bn1_b    = (const float*)d_in[6];
  const float* enc_w2   = (const float*)d_in[7];
  const float* enc_b2   = (const float*)d_in[8];
  const float* bn2_g    = (const float*)d_in[9];
  const float* bn2_b    = (const float*)d_in[10];
  const float* proj_w   = (const float*)d_in[11];
  const float* proj_b   = (const float*)d_in[12];
  const float* pln_g    = (const float*)d_in[13];
  const float* pln_b    = (const float*)d_in[14];
  const float* aln_g    = (const float*)d_in[15];
  const float* aln_b    = (const float*)d_in[16];
  const float* wq       = (const float*)d_in[17];
  const float* wk       = (const float*)d_in[18];
  const float* wv       = (const float*)d_in[19];
  const float* wo       = (const float*)d_in[20];
  const float* wo_b     = (const float*)d_in[21];
  const float* fln_g    = (const float*)d_in[22];
  const float* fln_b    = (const float*)d_in[23];
  const float* ff_w1    = (const float*)d_in[24];
  const float* ff_b1    = (const float*)d_in[25];
  const float* ff_w2    = (const float*)d_in[26];
  const float* ff_b2    = (const float*)d_in[27];
  const float* tok_c    = (const float*)d_in[28];
  const float* tok_n1   = (const float*)d_in[29];
  const float* tok_n2   = (const float*)d_in[30];
  const float* ctok     = (const float*)d_in[31];
  const float* pred_w1  = (const float*)d_in[32];
  const float* pred_b1  = (const float*)d_in[33];
  const float* pbn_g    = (const float*)d_in[34];
  const float* pbn_b    = (const float*)d_in[35];
  const float* pred_w2  = (const float*)d_in[36];
  float* out = (float*)d_out;

  // ---- workspace layout ----
  char* ws = (char*)d_ws;
  size_t off = 0;
  auto alloc = [&](size_t bytes) { off = (off + 255) & ~(size_t)255; size_t o = off; off += bytes; return o; };
  _Float16* w1h    = (_Float16*)(ws + alloc((size_t)896 * 512 * 2));
  _Float16* w2h    = (_Float16*)(ws + alloc((size_t)512 * 256 * 2));
  _Float16* projh  = (_Float16*)(ws + alloc((size_t)256 * 256 * 2));
  _Float16* wqh    = (_Float16*)(ws + alloc((size_t)256 * 256 * 2));
  _Float16* wkh    = (_Float16*)(ws + alloc((size_t)256 * 256 * 2));
  _Float16* wvh    = (_Float16*)(ws + alloc((size_t)256 * 256 * 2));
  _Float16* woh    = (_Float16*)(ws + alloc((size_t)256 * 256 * 2));
  _Float16* ff1h   = (_Float16*)(ws + alloc((size_t)256 * 512 * 2));
  _Float16* ff2h   = (_Float16*)(ws + alloc((size_t)512 * 256 * 2));
  _Float16* predWh = (_Float16*)(ws + alloc((size_t)256 * 17536 * 2));
  float*    valid  = (float*)(ws + alloc(R * 4));
  _Float16* xin16  = (_Float16*)(ws + alloc(R * 256 * 2));
  float*    fo     = (float*)(ws + alloc(R * 256 * 4));
  _Float16* feat16 = (_Float16*)(ws + alloc((size_t)Bb * 256 * 2));
  size_t arena = alloc(0);
  _Float16* xh    = (_Float16*)(ws + alloc(R * 896 * 2));
  float*    t32a  = (float*)(ws + alloc(R * 512 * 4));
  float*    t32b  = (float*)(ws + alloc(R * 256 * 4));
  float*    addt  = (float*)(ws + alloc(R * 256 * 4));
  _Float16* h1h   = (_Float16*)(ws + alloc(R * 512 * 2));   // also reused as FFN mid16
  float*    q32   = (float*)(ws + alloc(R * 256 * 4));
  float*    k32   = (float*)(ws + alloc(R * 256 * 4));
  float*    v32   = (float*)(ws + alloc(R * 256 * 4));
  _Float16* o16   = (_Float16*)(ws + alloc(R * 256 * 2));
  float*    ph    = (float*)(ws + arena);   // 287MB, overlaps dead early-stage buffers

  auto blocks = [](size_t n) { return (unsigned)((n + 255) / 256); };

  // weights -> f16
  conv_w1   <<<blocks((size_t)896 * 512), 256, 0, stream>>>(enc_w1, w1h);
  conv_plain<<<blocks((size_t)512 * 256), 256, 0, stream>>>(enc_w2, w2h, (size_t)512 * 256);
  conv_plain<<<blocks((size_t)256 * 256), 256, 0, stream>>>(proj_w, projh, (size_t)256 * 256);
  conv_plain<<<blocks((size_t)256 * 256), 256, 0, stream>>>(wq, wqh, (size_t)256 * 256);
  conv_plain<<<blocks((size_t)256 * 256), 256, 0, stream>>>(wk, wkh, (size_t)256 * 256);
  conv_plain<<<blocks((size_t)256 * 256), 256, 0, stream>>>(wv, wvh, (size_t)256 * 256);
  conv_plain<<<blocks((size_t)256 * 256), 256, 0, stream>>>(wo, woh, (size_t)256 * 256);
  conv_plain<<<blocks((size_t)256 * 512), 256, 0, stream>>>(ff_w1, ff1h, (size_t)256 * 512);
  conv_plain<<<blocks((size_t)512 * 256), 256, 0, stream>>>(ff_w2, ff2h, (size_t)512 * 256);
  conv_predw<<<blocks((size_t)256 * 17536), 256, 0, stream>>>(pred_w1, predWh);

  // inputs
  prep_x     <<<(unsigned)R, 256, 0, stream>>>(source, srcn, xh, valid);
  prep_tokens<<<(unsigned)R, 256, 0, stream>>>(cell_inf, tok_c, tok_n1, tok_n2, ctok, valid, addt);

  // encoder
  gemm16<<<dim3(512 / 64, (unsigned)(R / 64)), 128, 0, stream>>>(xh, w1h, t32a, (int)R, 512, 896, 896, 512, 512);
  ep_bn_leaky<<<blocks(R * 512), 256, 0, stream>>>(t32a, enc_b1, bn1_g, bn1_b, h1h, 512, R * 512);
  gemm16<<<dim3(256 / 64, (unsigned)(R / 64)), 128, 0, stream>>>(h1h, w2h, t32b, (int)R, 256, 512, 512, 256, 256);
  ep_bn_leaky_add<<<blocks(R * 256), 256, 0, stream>>>(t32b, enc_b2, bn2_g, bn2_b, addt, xin16, 256, R * 256);

  // projection + LN/relu + attn-LN
  gemm16<<<dim3(256 / 64, (unsigned)(R / 64)), 128, 0, stream>>>(xin16, projh, t32b, (int)R, 256, 256, 256, 256, 256);
  ep_ln_relu_ln<<<(unsigned)(R / 8), 256, 0, stream>>>(t32b, proj_b, pln_g, pln_b, aln_g, aln_b, fo, xin16, R);

  // attention
  gemm16<<<dim3(256 / 64, (unsigned)(R / 64)), 128, 0, stream>>>(xin16, wqh, q32, (int)R, 256, 256, 256, 256, 256);
  gemm16<<<dim3(256 / 64, (unsigned)(R / 64)), 128, 0, stream>>>(xin16, wkh, k32, (int)R, 256, 256, 256, 256, 256);
  gemm16<<<dim3(256 / 64, (unsigned)(R / 64)), 128, 0, stream>>>(xin16, wvh, v32, (int)R, 256, 256, 256, 256, 256);
  attn_kernel<<<Bb, 128, 0, stream>>>(q32, k32, v32, valid, o16);
  gemm16<<<dim3(256 / 64, (unsigned)(R / 64)), 128, 0, stream>>>(o16, woh, t32b, (int)R, 256, 256, 256, 256, 256);
  ep_res_ln<<<(unsigned)(R / 8), 256, 0, stream>>>(t32b, wo_b, fln_g, fln_b, fo, xin16, R);

  // FFN
  gemm16<<<dim3(512 / 64, (unsigned)(R / 64)), 128, 0, stream>>>(xin16, ff1h, t32a, (int)R, 512, 256, 256, 512, 512);
  ep_gelu<<<blocks(R * 512), 256, 0, stream>>>(t32a, ff_b1, h1h, 512, R * 512);
  gemm16<<<dim3(256 / 64, (unsigned)(R / 64)), 128, 0, stream>>>(h1h, ff2h, t32b, (int)R, 256, 512, 512, 256, 256);
  ep_res_add<<<blocks(R * 256), 256, 0, stream>>>(t32b, ff_b2, fo, 256, R * 256);

  // prediction head
  extract_feat<<<blocks((size_t)Bb * 256), 256, 0, stream>>>(fo, feat16, Bb);
  gemm16<<<dim3(17536 / 64, Bb / 64), 128, 0, stream>>>(feat16, predWh, ph, Bb, 17536, 256, 256, 17536, 17536);
  pred_reduce<<<(unsigned)(((size_t)Bb * T + 7) / 8), 256, 0, stream>>>(ph, pred_b1, pbn_g, pbn_b, pred_w2, out, Bb * T);
}
